// mSwin_18245021073658
// MI455X (gfx1250) — matrix operand order repaired and checked by reading, not compiled, not run on hardware
//
#include <hip/hip_runtime.h>
#include <hip/hip_bf16.h>
#include <math.h>

// ---------------------------------------------------------------------------
// Types for CDNA5 WMMA (wave32, 16x16x32 bf16 -> f32 accumulate)
// ---------------------------------------------------------------------------
typedef __bf16 bf16_t;
typedef __attribute__((ext_vector_type(16))) __bf16 v16bf;
typedef __attribute__((ext_vector_type(8)))  float  v8f;

#define DEV __device__ __forceinline__

// Model constants
#define NB     2
#define HH     240
#define WW     320
#define HWPIX  76800          // 240*320
#define CC     128
#define NHEADS 4
#define HD     32
#define NWX    46             // ceil(320/7)
#define NWY    35             // ceil(240/7)
#define NWIN   (NB*NWY*NWX)   // 3220
#define NTOK   (NB*HWPIX)     // 153600
#define HID    512
#define DOUT   256
#define NQ     128

// CDNA5 async global->LDS copy path (guarded: falls back to plain loads)
#if defined(__HIP_DEVICE_COMPILE__) && __has_builtin(__builtin_amdgcn_global_load_async_to_lds_b128)
#define HAVE_ASYNC 1
#else
#define HAVE_ASYNC 0
#endif

#if HAVE_ASYNC
typedef __attribute__((ext_vector_type(4))) int v4i_t;
DEV void async_copy_b128(const float* src, float* lds_dst) {
  __builtin_amdgcn_global_load_async_to_lds_b128(
      (__attribute__((address_space(1))) v4i_t*)(void*)const_cast<float*>(src),
      (__attribute__((address_space(3))) v4i_t*)(void*)lds_dst,
      0, 0);
}
DEV void wait_async0() {
#if __has_builtin(__builtin_amdgcn_s_wait_asynccnt)
  __builtin_amdgcn_s_wait_asynccnt(0);
#else
  asm volatile("s_wait_asynccnt 0x0" ::: "memory");
#endif
}
#endif

DEV v8f wmma_bf16(v16bf a, v16bf b, v8f c) {
  return __builtin_amdgcn_wmma_f32_16x16x32_bf16(false, a, false, b, (short)0, c,
                                                 false, false);
}

// A fragment: 16x32 tile, row-major source, lda elements.
// Lanes 0-15: row = lane, K = 0..15 ; lanes 16-31: K = 16..31.
DEV v16bf load_a_frag(const bf16_t* A, int lda, int lane) {
  const int row = lane & 15;
  const int kb  = (lane >> 4) << 4;     // 0 or 16
  const bf16_t* p = A + row * lda + kb;
  v16bf a;
#pragma unroll
  for (int i = 0; i < 16; ++i) a[i] = p[i];
  return a;
}

// B fragment: 32x16 tile of B where we hold W with W[n][k] = B[k][n]
// (i.e. weights stored (out,in) row-major). Each lane: 16 contiguous bf16.
DEV v16bf load_b_frag(const bf16_t* W, int ldw, int lane) {
  const int n  = lane & 15;
  const int kb = (lane >> 4) << 4;      // 0 or 16
  const bf16_t* p = W + n * ldw + kb;
  v16bf b;
#pragma unroll
  for (int i = 0; i < 16; ++i) b[i] = p[i];
  return b;
}

// Row-tile GEMM: one wave computes a 16-row stripe against JT 16-col weight
// tiles with KS k-steps of 32. B fragments stream in chunks of up to 4 tiles
// (distinct register arrays) so a chunk of loads is always in flight while the
// previous chunk feeds back-to-back WMMAs (partial loadcnt waits).
template <int JT, int KS>
DEV void gemm_rowtile(const bf16_t* A, int lda, const bf16_t* W, int ldw,
                      int lane, v8f* acc) {
  constexpr int CH = (JT % 4 == 0) ? 4 : ((JT % 3 == 0) ? 3 : ((JT % 2 == 0) ? 2 : 1));
  constexpr int NG = JT / CH;
#pragma unroll
  for (int ks = 0; ks < KS; ++ks) {
    v16bf a = load_a_frag(A + ks * 32, lda, lane);
    v16bf cur[CH];
#pragma unroll
    for (int c = 0; c < CH; ++c)
      cur[c] = load_b_frag(W + (size_t)c * 16 * ldw + ks * 32, ldw, lane);
#pragma unroll
    for (int g = 0; g < NG; ++g) {
      v16bf nxt[CH];
      if (g + 1 < NG) {
#pragma unroll
        for (int c = 0; c < CH; ++c)
          nxt[c] = load_b_frag(W + (size_t)((g + 1) * CH + c) * 16 * ldw + ks * 32,
                               ldw, lane);
      }
#pragma unroll
      for (int c = 0; c < CH; ++c)
        acc[g * CH + c] = wmma_bf16(a, cur[c], acc[g * CH + c]);
      if (g + 1 < NG) {
#pragma unroll
        for (int c = 0; c < CH; ++c) cur[c] = nxt[c];
      }
    }
  }
}

// Prefetch a bf16 weight panel (n elements) into cache, one line per thread.
DEV void prefetch_panel(const bf16_t* W, int nelem, int tid) {
  for (int off = tid * 64; off < nelem; off += 256 * 64)
    __builtin_prefetch(W + off, 0, 2);
}

DEV float wave_sum(float v) {
#pragma unroll
  for (int m = 16; m >= 1; m >>= 1) v += __shfl_xor(v, m, 32);
  return v;
}

// LayerNorm one 128-wide row with a full wave; write bf16 to dst.
DEV void ln_row_bf16(const float* src, const float* g, const float* b,
                     bf16_t* dst, int lane) {
  const int c0 = lane * 4;
  float v0 = src[c0], v1 = src[c0 + 1], v2 = src[c0 + 2], v3 = src[c0 + 3];
  float mu = wave_sum(v0 + v1 + v2 + v3) * (1.0f / 128.0f);
  float d0 = v0 - mu, d1 = v1 - mu, d2 = v2 - mu, d3 = v3 - mu;
  float var = wave_sum(d0 * d0 + d1 * d1 + d2 * d2 + d3 * d3) * (1.0f / 128.0f);
  float r = rsqrtf(var + 1e-5f);
  dst[c0]     = (bf16_t)(d0 * r * g[c0]     + b[c0]);
  dst[c0 + 1] = (bf16_t)(d1 * r * g[c0 + 1] + b[c0 + 1]);
  dst[c0 + 2] = (bf16_t)(d2 * r * g[c0 + 2] + b[c0 + 2]);
  dst[c0 + 3] = (bf16_t)(d3 * r * g[c0 + 3] + b[c0 + 3]);
}

// ---------------------------------------------------------------------------
// fp32 -> bf16 convert
// ---------------------------------------------------------------------------
__global__ void cvt_bf16_kernel(const float* __restrict__ s, bf16_t* __restrict__ d,
                                int n) {
  int i = blockIdx.x * 256 + threadIdx.x;
  if (i < n) d[i] = (bf16_t)s[i];
}

// ---------------------------------------------------------------------------
// Transpose x: (N,CIN,HW) f32 -> (N,HW,CIN) bf16.
// LDS tile [c][p] with 68-float (16B-aligned) pitch; phase 1 uses CDNA5
// GLOBAL_LOAD_ASYNC_TO_LDS_B128 when the toolchain exposes it.
// ---------------------------------------------------------------------------
__global__ __launch_bounds__(256) void xpose_kernel(const float* __restrict__ x,
                                                    bf16_t* __restrict__ xbf) {
  __shared__ float tile[128 * 68];   // 34816 B
  const int t = threadIdx.x;
  const int n = blockIdx.x / 1200;
  const size_t p0 = (size_t)(blockIdx.x % 1200) * 64;
  const float* xs = x + (size_t)n * CC * HWPIX + p0;
#if HAVE_ASYNC
#pragma unroll
  for (int it = 0; it < 8; ++it) {
    const int idx = it * 256 + t;           // 0..2047 b128 transfers
    const int c = idx >> 4, po = (idx & 15) * 4;
    async_copy_b128(xs + (size_t)c * HWPIX + po, &tile[c * 68 + po]);
  }
  wait_async0();
#else
  for (int idx = t; idx < 128 * 64; idx += 256) {
    const int c = idx >> 6, pl = idx & 63;
    tile[c * 68 + pl] = xs[(size_t)c * HWPIX + pl];
  }
#endif
  __syncthreads();
  bf16_t* d = xbf + ((size_t)n * HWPIX + p0) * CC;
  for (int idx = t; idx < 64 * 128; idx += 256) {
    const int pl = idx >> 7, c = idx & 127;
    d[(size_t)pl * CC + c] = (bf16_t)tile[c * 68 + pl];
  }
}

// ---------------------------------------------------------------------------
// 1x1 conv: tokens[m][e] = sum_c xbf[m][c] * conv_w[e][c] + b[e]
// 128 rows x 128 cols per workgroup (8 waves, wave = one 16-row stripe)
// ---------------------------------------------------------------------------
__global__ __launch_bounds__(256) void conv_kernel(
    const bf16_t* __restrict__ xbf, const bf16_t* __restrict__ wb,
    const float* __restrict__ bias, float* __restrict__ tok,
    bf16_t* __restrict__ tok0) {
  const int tid = threadIdx.x, lane = tid & 31, wv = tid >> 5;
  const size_t base = (size_t)blockIdx.x * 128;
  const bf16_t* A = xbf + (base + (size_t)wv * 16) * CC;
  prefetch_panel(wb, CC * CC, tid);
  v8f acc[8];
  const v8f z = {0.f, 0.f, 0.f, 0.f, 0.f, 0.f, 0.f, 0.f};
#pragma unroll
  for (int j = 0; j < 8; ++j) acc[j] = z;
  gemm_rowtile<8, 4>(A, CC, wb, CC, lane, acc);
#pragma unroll
  for (int j = 0; j < 8; ++j) {
    const int col = j * 16 + (lane & 15);
    const float bv = bias[col];
#pragma unroll
    for (int i = 0; i < 8; ++i) {
      const size_t m = base + wv * 16 + i + ((lane >> 4) << 3);
      const float v = acc[j][i] + bv;
      tok[m * CC + col] = v;
      tok0[m * CC + col] = (bf16_t)v;
    }
  }
}

// ---------------------------------------------------------------------------
// Fused window attention: LN1 + QKV + softmax(QK^T/sqrt(d)) V + proj + residual
// One workgroup (8 waves) per 7x7 window; windows padded to 64 rows in LDS.
// LDS (128 KB): t_bf[64][128] | q_l[4][64][32] | k_l[4][64][32] | vT[4][32][64]
//               | s_l[4][64][64] f32 ; p_bf aliases (t_bf,q_l); o_l aliases s_l
// ---------------------------------------------------------------------------
__global__ __launch_bounds__(256) void attn_kernel(
    float* __restrict__ tok, const bf16_t* __restrict__ qkv_wb,
    const float* __restrict__ qkv_b, const bf16_t* __restrict__ proj_wb,
    const float* __restrict__ proj_b, const float* __restrict__ n1_g,
    const float* __restrict__ n1_b) {
  extern __shared__ char smem[];
  bf16_t* t_bf = (bf16_t*)smem;              // 64*128   (16 KB)
  bf16_t* q_l  = t_bf + 64 * 128;            // 4*64*32  (16 KB)
  bf16_t* k_l  = q_l + 4 * 64 * 32;          // 4*64*32  (16 KB)
  bf16_t* vT   = k_l + 4 * 64 * 32;          // 4*32*64  (16 KB)
  float*  s_l  = (float*)(vT + 4 * 32 * 64); // 4*64*64 f32 (64 KB)
  bf16_t* p_bf = t_bf;                       // alias over t_bf+q_l (32 KB)
  bf16_t* o_l  = (bf16_t*)s_l;               // alias over s_l (16 KB)

  const int tid = threadIdx.x, lane = tid & 31, wv = tid >> 5;
  const int blk = blockIdx.x;
  const int n = blk / (NWY * NWX);
  const int rem = blk % (NWY * NWX);
  const int wy = rem / NWX, wx = rem % NWX;

  prefetch_panel(qkv_wb, 384 * CC, tid);

  // ---- 1. LN1 into windowed bf16 tile (zeros for padded / fake rows) ----
  for (int r = wv; r < 64; r += 8) {
    bool valid = false; int hy = 0, hx = 0;
    if (r < 49) { hy = wy * 7 + r / 7; hx = wx * 7 + r % 7; valid = (hy < HH) && (hx < WW); }
    if (valid) {
      const float* src = tok + ((size_t)n * HWPIX + (size_t)hy * WW + hx) * CC;
      ln_row_bf16(src, n1_g, n1_b, t_bf + r * 128, lane);
    } else {
      const int c0 = lane * 4;
#pragma unroll
      for (int i = 0; i < 4; ++i) t_bf[r * 128 + c0 + i] = (bf16_t)0.0f;
    }
  }
  __syncthreads();

  // ---- 2. QKV GEMM: 64x384 = 4x24 tiles, 12 per wave ----
  {
    const int m_t = wv >> 1;
    const int nbase = (wv & 1) * 12;
    v8f acc[12];
    const v8f z = {0.f, 0.f, 0.f, 0.f, 0.f, 0.f, 0.f, 0.f};
#pragma unroll
    for (int j = 0; j < 12; ++j) acc[j] = z;
    gemm_rowtile<12, 4>(t_bf + m_t * 16 * 128, 128,
                        qkv_wb + (size_t)nbase * 16 * CC, CC, lane, acc);
#pragma unroll
    for (int j = 0; j < 12; ++j) {
      const int colg = (nbase + j) * 16 + (lane & 15);
      const float bv = qkv_b[colg];
#pragma unroll
      for (int i = 0; i < 8; ++i) {
        const int m = m_t * 16 + i + ((lane >> 4) << 3);
        const float val = acc[j][i] + bv;
        if (colg < 128) {
          const int h = colg >> 5, d = colg & 31;
          q_l[(h * 64 + m) * 32 + d] = (bf16_t)val;
        } else if (colg < 256) {
          const int h = (colg - 128) >> 5, d = colg & 31;
          k_l[(h * 64 + m) * 32 + d] = (bf16_t)val;
        } else {
          const int h = (colg - 256) >> 5, d = colg & 31;
          vT[(h * 32 + d) * 64 + m] = (bf16_t)val;   // store V transposed
        }
      }
    }
  }
  __syncthreads();

  // ---- 3. Scores S = Q K^T * scale : per head 64x64, 8 tiles per wave ----
  {
    const int h = wv >> 1;
    const bf16_t* qh = q_l + h * 64 * 32;
    const bf16_t* kh = k_l + h * 64 * 32;
    float* sh = s_l + h * 64 * 64;
#pragma unroll
    for (int m_t = 0; m_t < 4; ++m_t) {
      v16bf a = load_a_frag(qh + m_t * 16 * 32, 32, lane);
#pragma unroll
      for (int jj = 0; jj < 2; ++jj) {
        const int n_t = (wv & 1) * 2 + jj;
        v16bf b = load_b_frag(kh + n_t * 16 * 32, 32, lane);
        v8f acc = {0.f, 0.f, 0.f, 0.f, 0.f, 0.f, 0.f, 0.f};
        acc = wmma_bf16(a, b, acc);
        const int col = n_t * 16 + (lane & 15);
#pragma unroll
        for (int i = 0; i < 8; ++i) {
          const int m = m_t * 16 + i + ((lane >> 4) << 3);
          sh[m * 64 + col] = acc[i] * 0.17677669529663687f;  // 1/sqrt(32)
        }
      }
    }
  }
  __syncthreads();

  // ---- 4. Softmax over the 49 real key positions (one thread per row) ----
  {
    const int h = tid >> 6, r = tid & 63;
    float* row = s_l + (h * 64 + r) * 64;
    float mx = -1e30f;
    for (int k = 0; k < 49; ++k) mx = fmaxf(mx, row[k]);
    float sum = 0.f;
    for (int k = 0; k < 49; ++k) { float e = __expf(row[k] - mx); row[k] = e; sum += e; }
    const float inv = 1.0f / sum;
    bf16_t* prow = p_bf + (h * 64 + r) * 64;
    for (int k = 0; k < 49; ++k) prow[k] = (bf16_t)(row[k] * inv);
    for (int k = 49; k < 64; ++k) prow[k] = (bf16_t)0.0f;
  }
  __syncthreads();

  // ---- 5. O = P V : per head 64x32 (4x2 tiles), 4 tiles per wave ----
  {
    const int h = wv >> 1, n_t = wv & 1;
    const bf16_t* ph = p_bf + h * 64 * 64;
    const bf16_t* vh = vT + h * 32 * 64;
#pragma unroll
    for (int m_t = 0; m_t < 4; ++m_t) {
      v8f acc = {0.f, 0.f, 0.f, 0.f, 0.f, 0.f, 0.f, 0.f};
#pragma unroll
      for (int ks = 0; ks < 2; ++ks) {
        v16bf a = load_a_frag(ph + m_t * 16 * 64 + ks * 32, 64, lane);
        v16bf b = load_b_frag(vh + n_t * 16 * 64 + ks * 32, 64, lane);
        acc = wmma_bf16(a, b, acc);
      }
      const int d = n_t * 16 + (lane & 15);
#pragma unroll
      for (int i = 0; i < 8; ++i) {
        const int m = m_t * 16 + i + ((lane >> 4) << 3);
        o_l[m * 128 + h * 32 + d] = (bf16_t)acc[i];
      }
    }
  }
  __syncthreads();

  // ---- 6. proj + residual back to global tokens ----
  {
    const int m_t = wv >> 1;
    const int nb = (wv & 1) * 4;
    v8f acc[4];
    const v8f z = {0.f, 0.f, 0.f, 0.f, 0.f, 0.f, 0.f, 0.f};
#pragma unroll
    for (int j = 0; j < 4; ++j) acc[j] = z;
    gemm_rowtile<4, 4>(o_l + m_t * 16 * 128, 128,
                       proj_wb + (size_t)nb * 16 * CC, CC, lane, acc);
#pragma unroll
    for (int j = 0; j < 4; ++j) {
      const int col = (nb + j) * 16 + (lane & 15);
      const float bv = proj_b[col];
#pragma unroll
      for (int i = 0; i < 8; ++i) {
        const int m = m_t * 16 + i + ((lane >> 4) << 3);
        if (m < 49) {
          const int hy = wy * 7 + m / 7, hx = wx * 7 + m % 7;
          if (hy < HH && hx < WW) {
            const size_t idx = ((size_t)n * HWPIX + (size_t)hy * WW + hx) * CC + col;
            tok[idx] += acc[j][i] + bv;
          }
        }
      }
    }
  }
}

// ---------------------------------------------------------------------------
// Fused MLP: LN2 -> fc1 (+GELU) -> fc2 (+bias+residual). 64 rows per WG.
// LDS: h_bf[64][128] bf16 (16 KB) + m_bf[64][512] bf16 (64 KB)
// fc1 runs as two 8-tile halves to cap live accumulator VGPRs.
// ---------------------------------------------------------------------------
__global__ __launch_bounds__(256) void mlp_kernel(
    float* __restrict__ tok, const bf16_t* __restrict__ fc1_wb,
    const float* __restrict__ fc1_b, const bf16_t* __restrict__ fc2_wb,
    const float* __restrict__ fc2_b, const float* __restrict__ n2_g,
    const float* __restrict__ n2_b) {
  extern __shared__ char smem[];
  bf16_t* h_bf = (bf16_t*)smem;         // 64*128
  bf16_t* m_bf = h_bf + 64 * 128;       // 64*512
  const int tid = threadIdx.x, lane = tid & 31, wv = tid >> 5;
  const size_t base = (size_t)blockIdx.x * 64;
  const v8f z = {0.f, 0.f, 0.f, 0.f, 0.f, 0.f, 0.f, 0.f};

  prefetch_panel(fc1_wb, HID * CC, tid);

  for (int r = wv; r < 64; r += 8)
    ln_row_bf16(tok + (base + r) * CC, n2_g, n2_b, h_bf + r * 128, lane);
  __syncthreads();

  // fc1: 64x512 = 4x32 tiles; wave covers 16 tiles in two halves of 8
  {
    const int m_t = wv >> 1;
#pragma unroll
    for (int half = 0; half < 2; ++half) {
      const int nb = (wv & 1) * 16 + half * 8;
      v8f acc[8];
#pragma unroll
      for (int j = 0; j < 8; ++j) acc[j] = z;
      gemm_rowtile<8, 4>(h_bf + m_t * 16 * 128, 128,
                         fc1_wb + (size_t)nb * 16 * CC, CC, lane, acc);
#pragma unroll
      for (int j = 0; j < 8; ++j) {
        const int col = (nb + j) * 16 + (lane & 15);
        const float bv = fc1_b[col];
#pragma unroll
        for (int i = 0; i < 8; ++i) {
          const int m = m_t * 16 + i + ((lane >> 4) << 3);
          const float x = acc[j][i] + bv;
          const float g = 0.5f * x * (1.0f + erff(x * 0.7071067811865475f)); // exact GELU
          m_bf[m * HID + col] = (bf16_t)g;
        }
      }
    }
  }
  prefetch_panel(fc2_wb, CC * HID, tid);
  __syncthreads();

  // fc2: 64x128 = 4x8 tiles, 4 per wave, K = 512
  {
    const int m_t = wv >> 1;
    const int nb = (wv & 1) * 4;
    v8f acc[4];
#pragma unroll
    for (int j = 0; j < 4; ++j) acc[j] = z;
    gemm_rowtile<4, 16>(m_bf + m_t * 16 * HID, HID,
                        fc2_wb + (size_t)nb * 16 * HID, HID, lane, acc);
#pragma unroll
    for (int j = 0; j < 4; ++j) {
      const int col = (nb + j) * 16 + (lane & 15);
      const float bv = fc2_b[col];
#pragma unroll
      for (int i = 0; i < 8; ++i) {
        const int m = m_t * 16 + i + ((lane >> 4) << 3);
        const size_t idx = (base + m) * CC + col;
        tok[idx] = tok[idx] + acc[j][i] + bv;
      }
    }
  }
}

// ---------------------------------------------------------------------------
// head mean: head[n*128+c] = mean_p tok[n][p][c]   (256 blocks, one per (n,c))
// ---------------------------------------------------------------------------
__global__ void head_mean_kernel(const float* __restrict__ tok,
                                 float* __restrict__ head) {
  __shared__ float red[256];
  const int b = blockIdx.x, t = threadIdx.x;
  const int n = b >> 7, c = b & 127;
  const float* base = tok + (size_t)n * HWPIX * CC + c;
  float s = 0.f;
  for (int p = t; p < HWPIX; p += 256) s += base[(size_t)p * CC];
  red[t] = s;
  __syncthreads();
  for (int m = 128; m >= 1; m >>= 1) {
    if (t < m) red[t] += red[t + m];
    __syncthreads();
  }
  if (t == 0) head[b] = red[0] * (1.0f / (float)HWPIX);
}

// ---------------------------------------------------------------------------
// head MLP: leaky(128->256), leaky(256->256), 256->256, relu+0.1, normalize
// ---------------------------------------------------------------------------
__global__ void head_mlp_kernel(const float* __restrict__ head,
                                const float* __restrict__ r1_w, const float* __restrict__ r1_b,
                                const float* __restrict__ r2_w, const float* __restrict__ r2_b,
                                const float* __restrict__ r3_w, const float* __restrict__ r3_b,
                                float* __restrict__ y) {
  __shared__ float h[128], y1[256], y2[256], y3[256], red[256];
  const int n = blockIdx.x, t = threadIdx.x;
  if (t < 128) h[t] = head[n * 128 + t];
  __syncthreads();
  float a = r1_b[t];
  for (int k = 0; k < 128; ++k) a += r1_w[t * 128 + k] * h[k];
  y1[t] = (a > 0.f) ? a : 0.01f * a;
  __syncthreads();
  float b2 = r2_b[t];
  for (int k = 0; k < 256; ++k) b2 += r2_w[t * 256 + k] * y1[k];
  y2[t] = (b2 > 0.f) ? b2 : 0.01f * b2;
  __syncthreads();
  float c3 = r3_b[t];
  for (int k = 0; k < 256; ++k) c3 += r3_w[t * 256 + k] * y2[k];
  c3 = fmaxf(c3, 0.f) + 0.1f;
  y3[t] = c3;
  red[t] = c3;
  __syncthreads();
  for (int m = 128; m >= 1; m >>= 1) {
    if (t < m) red[t] += red[t + m];
    __syncthreads();
  }
  __syncthreads();
  y[n * DOUT + t] = y3[t] / red[0];
}

// ---------------------------------------------------------------------------
// queries = final tokens[:, :128, :] -> bf16
// ---------------------------------------------------------------------------
__global__ void queries_kernel(const float* __restrict__ tok,
                               bf16_t* __restrict__ qbf) {
  const int i = blockIdx.x * 256 + threadIdx.x;
  if (i >= NB * NQ * CC) return;
  const int n = i / (NQ * CC), r = i % (NQ * CC);
  qbf[i] = (bf16_t)tok[((size_t)n * HWPIX + (r >> 7)) * CC + (r & 127)];
}

// ---------------------------------------------------------------------------
// maps[n][q][p] = sum_c tok0[n][p][c] * qbf[n][q][c]
// WMMA GEMM, then LDS-staged fully coalesced 512B-row stores (biggest HBM
// write of the net: 78.6 MB -> must be bandwidth-clean at 23.3 TB/s).
// ---------------------------------------------------------------------------
__global__ __launch_bounds__(256) void maps_kernel(
    const bf16_t* __restrict__ tok0, const bf16_t* __restrict__ qbf,
    float* __restrict__ maps) {
  extern __shared__ char smem[];
  float* tile = (float*)smem;   // [128][129] f32 = 66048 B
  const int tid = threadIdx.x, lane = tid & 31, wv = tid >> 5;
  const int n = blockIdx.x / 600;
  const size_t p0 = (size_t)(blockIdx.x % 600) * 128;
  const bf16_t* A = tok0 + ((size_t)n * HWPIX + p0 + (size_t)wv * 16) * CC;
  const bf16_t* Wq = qbf + (size_t)n * NQ * CC;
  prefetch_panel(Wq, NQ * CC, tid);
  v8f acc[8];
  const v8f z = {0.f, 0.f, 0.f, 0.f, 0.f, 0.f, 0.f, 0.f};
#pragma unroll
  for (int j = 0; j < 8; ++j) acc[j] = z;
  gemm_rowtile<8, 4>(A, CC, Wq, CC, lane, acc);
  // stage tile[q][p_local] in LDS (padded rows: conflict-free both phases)
#pragma unroll
  for (int j = 0; j < 8; ++j) {
    const int q = j * 16 + (lane & 15);
#pragma unroll
    for (int i = 0; i < 8; ++i) {
      const int pl = wv * 16 + i + ((lane >> 4) << 3);
      tile[q * 129 + pl] = acc[j][i];
    }
  }
  __syncthreads();
  float* outb = maps + (size_t)n * NQ * HWPIX + p0;
  for (int idx = tid; idx < 128 * 128; idx += 256) {
    const int q = idx >> 7, pl = idx & 127;
    outb[(size_t)q * HWPIX + pl] = tile[q * 129 + pl];
  }
}

// ---------------------------------------------------------------------------
// Host launcher
// ---------------------------------------------------------------------------
extern "C" void kernel_launch(void* const* d_in, const int* in_sizes, int n_in,
                              void* d_out, int out_size, void* d_ws, size_t ws_size,
                              hipStream_t stream) {
  (void)in_sizes; (void)n_in; (void)out_size; (void)ws_size;
  const float* x      = (const float*)d_in[0];
  const float* conv_w = (const float*)d_in[1];
  const float* conv_b = (const float*)d_in[2];
  const float* n1_g   = (const float*)d_in[3];
  const float* n1_b   = (const float*)d_in[4];
  const float* qkv_w  = (const float*)d_in[5];
  const float* qkv_b  = (const float*)d_in[6];
  const float* proj_w = (const float*)d_in[7];
  const float* proj_b = (const float*)d_in[8];
  const float* n2_g   = (const float*)d_in[9];
  const float* n2_b   = (const float*)d_in[10];
  const float* fc1_w  = (const float*)d_in[11];
  const float* fc1_b  = (const float*)d_in[12];
  const float* fc2_w  = (const float*)d_in[13];
  const float* fc2_b  = (const float*)d_in[14];
  const float* r1_w   = (const float*)d_in[15];
  const float* r1_b   = (const float*)d_in[16];
  const float* r2_w   = (const float*)d_in[17];
  const float* r2_b   = (const float*)d_in[18];
  const float* r3_w   = (const float*)d_in[19];
  const float* r3_b   = (const float*)d_in[20];

  char* ws = (char*)d_ws;
  float*  tok     = (float*)ws;  ws += (size_t)NTOK * CC * sizeof(float);   // 78.6 MB
  bf16_t* tok0    = (bf16_t*)ws; ws += (size_t)NTOK * CC * sizeof(bf16_t);  // 39.3 MB
  bf16_t* xbf     = (bf16_t*)ws; ws += (size_t)NTOK * CC * sizeof(bf16_t);  // 39.3 MB
  bf16_t* conv_wb = (bf16_t*)ws; ws += (size_t)CC * CC * sizeof(bf16_t);
  bf16_t* qkv_wb  = (bf16_t*)ws; ws += (size_t)4 * 384 * CC * sizeof(bf16_t);
  bf16_t* proj_wb = (bf16_t*)ws; ws += (size_t)4 * CC * CC * sizeof(bf16_t);
  bf16_t* fc1_wb  = (bf16_t*)ws; ws += (size_t)4 * HID * CC * sizeof(bf16_t);
  bf16_t* fc2_wb  = (bf16_t*)ws; ws += (size_t)4 * CC * HID * sizeof(bf16_t);
  float*  headb   = (float*)ws;  ws += (size_t)NB * CC * sizeof(float);
  bf16_t* qbf     = (bf16_t*)ws; ws += (size_t)NB * NQ * CC * sizeof(bf16_t);

  float* y_out    = (float*)d_out;       // (2,256)
  float* maps_out = y_out + NB * DOUT;   // (2,128,240,320)

  // Weight conversions to bf16
  cvt_bf16_kernel<<<(CC * CC + 255) / 256, 256, 0, stream>>>(conv_w, conv_wb, CC * CC);
  cvt_bf16_kernel<<<(4 * 384 * CC + 255) / 256, 256, 0, stream>>>(qkv_w, qkv_wb, 4 * 384 * CC);
  cvt_bf16_kernel<<<(4 * CC * CC + 255) / 256, 256, 0, stream>>>(proj_w, proj_wb, 4 * CC * CC);
  cvt_bf16_kernel<<<(4 * HID * CC + 255) / 256, 256, 0, stream>>>(fc1_w, fc1_wb, 4 * HID * CC);
  cvt_bf16_kernel<<<(4 * CC * HID + 255) / 256, 256, 0, stream>>>(fc2_w, fc2_wb, 4 * CC * HID);

  // x -> (N,HW,C) bf16
  xpose_kernel<<<NB * 1200, 256, 0, stream>>>(x, xbf);

  // 1x1 conv -> tokens (fp32 residual stream) + tok0 (bf16, kept for maps)
  conv_kernel<<<NTOK / 128, 256, 0, stream>>>(xbf, conv_wb, conv_b, tok, tok0);

  // 4 transformer blocks
  const size_t ATTN_LDS = 131072;  // 128 KB (overlaid buffers)
  const size_t MLP_LDS  = 81920;   // 80 KB
  for (int d = 0; d < 4; ++d) {
    attn_kernel<<<NWIN, 256, ATTN_LDS, stream>>>(
        tok, qkv_wb + (size_t)d * 384 * CC, qkv_b + d * 384,
        proj_wb + (size_t)d * CC * CC, proj_b + d * CC,
        n1_g + d * CC, n1_b + d * CC);
    mlp_kernel<<<NTOK / 64, 256, MLP_LDS, stream>>>(
        tok, fc1_wb + (size_t)d * HID * CC, fc1_b + d * HID,
        fc2_wb + (size_t)d * CC * HID, fc2_b + d * CC,
        n2_g + d * CC, n2_b + d * CC);
  }

  // Head
  head_mean_kernel<<<NB * CC, 256, 0, stream>>>(tok, headb);
  head_mlp_kernel<<<NB, 256, 0, stream>>>(headb, r1_w, r1_b, r2_w, r2_b, r3_w, r3_b, y_out);

  // Maps
  queries_kernel<<<(NB * NQ * CC + 255) / 256, 256, 0, stream>>>(tok, qbf);
  maps_kernel<<<NB * 600, 256, 66048, stream>>>(tok0, qbf, maps_out);
}
